// TransformerBlock_90812788506838
// MI455X (gfx1250) — matrix operand order repaired and checked by reading, not compiled, not run on hardware
//
#include <hip/hip_runtime.h>
#include <hip/hip_bf16.h>

// ---------------------------------------------------------------------------
// Restormer-style transformer block for MI455X (gfx1250, wave32, WMMA).
// GEMMs (1x1 convs) run on v_wmma_f32_16x16x32_f16 with f32 accumulation;
// activations carried as f16 (23.3 TB/s HBM roofline), weights pre-converted
// to zero-padded f16, LDS staging via GLOBAL_LOAD_ASYNC_TO_LDS_B128
// (ASYNCcnt-tracked). All GEMM tile extents are compile-time multiples of the
// block tile so epilogues are branch-free.
// ---------------------------------------------------------------------------

typedef _Float16 h8   __attribute__((ext_vector_type(8)));
typedef _Float16 v16h __attribute__((ext_vector_type(16)));
typedef float    v8f  __attribute__((ext_vector_type(8)));

#define HW    16384
#define IMW   128
#define NPIX  65536   // B * HW, B = 4
#define CDIM  192
#define C3    576
#define HID   510
#define HID2  1020
#define H2P   1024    // padded FFN hidden channel count (GEMM stores zeros in pad)
#define HPAD  512

#if defined(__gfx1250__) && __has_builtin(__builtin_amdgcn_global_load_async_to_lds_b128)
#define USE_ASYNC_LDS 1
#else
#define USE_ASYNC_LDS 0
#endif

#if USE_ASYNC_LDS
// Builtin signature (from clang diagnostic): arg0 = 16B int-vector pointer in
// the global (__device__/AS1) address space, arg1 = LDS (AS3), then imm
// offset and cache policy.
typedef int v4i_ __attribute__((vector_size(4 * sizeof(int))));
typedef __attribute__((address_space(1))) v4i_ g_v4i;
typedef __attribute__((address_space(3))) v4i_ l_v4i;
__device__ __forceinline__ void async_cp16(const _Float16* g, _Float16* l) {
  __builtin_amdgcn_global_load_async_to_lds_b128((g_v4i*)g, (l_v4i*)l, 0, 0);
}
#endif

// ---------------------------------------------------------------------------
// Weight prep: fp32 (O x Creal) -> f16 zero-padded (Opad x Cpad).
// ---------------------------------------------------------------------------
__global__ __launch_bounds__(256)
void k_wcvt(const float* __restrict__ src, _Float16* __restrict__ dst,
            int O, int Creal, int Cpad, int total) {
  int idx = blockIdx.x * 256 + threadIdx.x;
  if (idx >= total) return;
  int o = idx / Cpad, c = idx - o * Cpad;
  float v = (o < O && c < Creal) ? src[(size_t)o * Creal + c] : 0.f;
  dst[idx] = (_Float16)v;
}

// ---------------------------------------------------------------------------
// LayerNorm over channel dim (NCHW, C=192) fused with transpose to
// pixel-major f16 (NPIX x 192) for the WMMA GEMM B-operand.
// ---------------------------------------------------------------------------
__global__ __launch_bounds__(256)
void k_ln_to_t(const float* __restrict__ xin, const float* __restrict__ w,
               const float* __restrict__ bia, _Float16* __restrict__ yt) {
  __shared__ float tile[CDIM * 65];
  __shared__ float part1[4 * 64];
  __shared__ float part2[4 * 64];
  __shared__ float muS[64];
  __shared__ float rsS[64];
  const int tid = threadIdx.x;
  const int pbase = blockIdx.x * 64;
  const int b = pbase >> 14;
  const int hw0 = pbase & (HW - 1);
  const float* src = xin + (size_t)b * CDIM * HW + hw0;
  for (int idx = tid; idx < CDIM * 64; idx += 256) {
    int c = idx >> 6, p = idx & 63;
    tile[c * 65 + p] = src[(size_t)c * HW + p];
  }
  __syncthreads();
  {
    int p = tid & 63, pa = tid >> 6;  // 4 threads per pixel
    float s = 0.f, s2 = 0.f;
    for (int c = pa; c < CDIM; c += 4) {
      float v = tile[c * 65 + p];
      s += v; s2 += v * v;
    }
    part1[pa * 64 + p] = s;
    part2[pa * 64 + p] = s2;
  }
  __syncthreads();
  if (tid < 64) {
    int p = tid;
    float S  = part1[p] + part1[64 + p] + part1[128 + p] + part1[192 + p];
    float S2 = part2[p] + part2[64 + p] + part2[128 + p] + part2[192 + p];
    float m = S * (1.f / CDIM);
    float var = S2 * (1.f / CDIM) - m * m;
    muS[p] = m;
    rsS[p] = 1.f / sqrtf(var + 1e-5f);
  }
  __syncthreads();
  for (int idx = tid; idx < 64 * CDIM; idx += 256) {
    int pp = idx / CDIM, c = idx - pp * CDIM;
    float v = (tile[c * 65 + pp] - muS[pp]) * rsS[pp] * w[c] + bia[c];
    yt[(size_t)(pbase + pp) * CDIM + c] = (_Float16)v;
  }
}

// ---------------------------------------------------------------------------
// WMMA GEMM: Out(OPAD x NPIX) = Wh(OPAD x KPAD f16) * Bt^T, Bt pixel-major f16
// (NPIX x KPAD). Block tile M=64, N=256, K=32; 8 waves in a 2x4 (M x N) grid,
// each wave owns a 32x64 tile = 2x4 v_wmma_f32_16x16x32_f16 accumulators.
// OPAD is a compile-time multiple of 64 -> branch-free epilogue.
// Epilogue: RESID=false -> f16 channel-major (B,OPAD,HW);
//           RESID=true  -> fp32 residual add into (B,OPAD,HW).
// ---------------------------------------------------------------------------
template <bool RESID, int KPAD, int OPAD>
__global__ __launch_bounds__(256)
void k_gemm_wmma(const _Float16* __restrict__ Wh, const _Float16* __restrict__ Bt,
                 _Float16* __restrict__ outH,
                 const float* __restrict__ res,
                 float* __restrict__ outF) {
  __shared__ __align__(16) _Float16 ldsA[64 * 40];    // 80B row stride (pad)
  __shared__ __align__(16) _Float16 ldsB[256 * 40];
  const int tid  = threadIdx.x;
  const int lane = tid & 31;
  const int wave = tid >> 5;
  const int wm = wave >> 2, wn = wave & 3;
  const int pBase = blockIdx.x * 256;
  const int oBase = blockIdx.y * 64;

  v8f acc[2][4];
  for (int i = 0; i < 2; ++i)
    for (int j = 0; j < 4; ++j)
      for (int e = 0; e < 8; ++e) acc[i][j][e] = 0.f;

  const int hi = lane >> 4;     // which half of the wave
  const int ml = lane & 15;

  // Per-thread staging coordinates (16B granules).
  const int ar = tid >> 2, aj8 = (tid & 3) * 8;           // A: 64x32 = 256 granules

  for (int k0 = 0; k0 < KPAD; k0 += 32) {
#if USE_ASYNC_LDS
    async_cp16(Wh + (size_t)(oBase + ar) * KPAD + k0 + aj8, &ldsA[ar * 40 + aj8]);
    for (int it = 0; it < 4; ++it) {
      int cc = tid + it * 256;
      int n = cc >> 2, j8 = (cc & 3) * 8;
      async_cp16(Bt + (size_t)(pBase + n) * KPAD + k0 + j8, &ldsB[n * 40 + j8]);
    }
    asm volatile("s_wait_asynccnt 0x0" ::: "memory");
#else
    {
      h8 va = *(const h8*)(Wh + (size_t)(oBase + ar) * KPAD + k0 + aj8);
      *(h8*)(&ldsA[ar * 40 + aj8]) = va;
      for (int it = 0; it < 4; ++it) {
        int cc = tid + it * 256;
        int n = cc >> 2, j8 = (cc & 3) * 8;
        h8 v = *(const h8*)(Bt + (size_t)(pBase + n) * KPAD + k0 + j8);
        *(h8*)(&ldsB[n * 40 + j8]) = v;
      }
    }
    if (k0 + 32 < KPAD)
      __builtin_prefetch(Bt + (size_t)(pBase + (tid >> 1)) * KPAD + k0 + 32, 0, 1);
#endif
    __syncthreads();

    v16h aF[2], bF[4];
    for (int i = 0; i < 2; ++i) {
      // A 16x32 f16 layout: lanes 0-15 hold K{0..7,16..23}, lanes 16-31 K{8..15,24..31}
      const _Float16* pA = ldsA + (wm * 32 + i * 16 + ml) * 40;
      h8 lo = *(const h8*)(pA + hi * 8);
      h8 hb = *(const h8*)(pA + 16 + hi * 8);
      aF[i] = __builtin_shufflevector(lo, hb, 0, 1, 2, 3, 4, 5, 6, 7,
                                      8, 9, 10, 11, 12, 13, 14, 15);
    }
    for (int j = 0; j < 4; ++j) {
      // B 32x16 f16: lanes 0-15 column n with K{0..7,16..23}, lanes 16-31 K{8..15,24..31}
      const _Float16* pB = ldsB + (wn * 64 + j * 16 + ml) * 40;
      h8 lo = *(const h8*)(pB + hi * 8);
      h8 hb = *(const h8*)(pB + 16 + hi * 8);
      bF[j] = __builtin_shufflevector(lo, hb, 0, 1, 2, 3, 4, 5, 6, 7,
                                      8, 9, 10, 11, 12, 13, 14, 15);
    }
    for (int i = 0; i < 2; ++i)
      for (int j = 0; j < 4; ++j)
        acc[i][j] = __builtin_amdgcn_wmma_f32_16x16x32_f16(
            false, aF[i], false, bF[j], (short)0, acc[i][j], false, false);
    __syncthreads();
  }

  // Epilogue: documented D layout — VGPR r, lanes 0-15 -> M=r, lanes 16-31 -> M=8+r.
  const int nl = lane & 15;
  const int rbase = (lane < 16) ? 0 : 8;
#pragma unroll
  for (int i = 0; i < 2; ++i)
#pragma unroll
    for (int j = 0; j < 4; ++j) {
      const int n = pBase + wn * 64 + j * 16 + nl;
      const int b = n >> 14, hw = n & (HW - 1);
#pragma unroll
      for (int r = 0; r < 8; ++r) {
        const int o = oBase + wm * 32 + i * 16 + rbase + r;
        const size_t gi = ((size_t)b * OPAD + o) * HW + hw;
        if (RESID) outF[gi] = res[gi] + acc[i][j][r];
        else       outH[gi] = (_Float16)acc[i][j][r];
      }
    }
}

// ---------------------------------------------------------------------------
// Depthwise 3x3 SAME conv, f16 in/out, fp32 weights + accumulation.
// ---------------------------------------------------------------------------
__global__ __launch_bounds__(256)
void k_dwconv(const _Float16* __restrict__ in, const float* __restrict__ w,
              _Float16* __restrict__ out, int Cx) {
  const int bc = blockIdx.x;                    // b*Cx + c
  const int hw = blockIdx.y * 256 + threadIdx.x;
  const int y = hw >> 7, x = hw & 127;
  const _Float16* p = in + (size_t)bc * HW;
  const float* wc = w + (bc % Cx) * 9;
  float acc = 0.f;
#pragma unroll
  for (int dy = -1; dy <= 1; ++dy) {
    int yy = y + dy;
    if (yy < 0 || yy > 127) continue;
#pragma unroll
    for (int dx = -1; dx <= 1; ++dx) {
      int xx = x + dx;
      if (xx < 0 || xx > 127) continue;
      acc += wc[(dy + 1) * 3 + (dx + 1)] * (float)p[yy * IMW + xx];
    }
  }
  out[(size_t)bc * HW + hw] = (_Float16)acc;
}

// Depthwise 3x3 on both FFN halves fused with exact GELU gating:
// g = gelu(conv(c)) * conv(c + HID). Input has padded channel stride H2P.
__global__ __launch_bounds__(256)
void k_dwconv_gate(const _Float16* __restrict__ in, const float* __restrict__ w,
                   _Float16* __restrict__ out) {
  const int bc = blockIdx.x;                    // b*HID + c
  const int b = bc / HID, c = bc - b * HID;
  const int hw = blockIdx.y * 256 + threadIdx.x;
  const int y = hw >> 7, x = hw & 127;
  const _Float16* p1 = in + ((size_t)b * H2P + c) * HW;
  const _Float16* p2 = in + ((size_t)b * H2P + HID + c) * HW;
  const float* w1 = w + c * 9;
  const float* w2 = w + (HID + c) * 9;
  float a1 = 0.f, a2 = 0.f;
#pragma unroll
  for (int dy = -1; dy <= 1; ++dy) {
    int yy = y + dy;
    if (yy < 0 || yy > 127) continue;
#pragma unroll
    for (int dx = -1; dx <= 1; ++dx) {
      int xx = x + dx;
      if (xx < 0 || xx > 127) continue;
      float wt1 = w1[(dy + 1) * 3 + (dx + 1)];
      float wt2 = w2[(dy + 1) * 3 + (dx + 1)];
      a1 += wt1 * (float)p1[yy * IMW + xx];
      a2 += wt2 * (float)p2[yy * IMW + xx];
    }
  }
  float g = 0.5f * a1 * (1.f + erff(a1 * 0.70710678118654752f));  // exact GELU
  out[((size_t)b * HID + c) * HW + hw] = (_Float16)(g * a2);
}

// ---------------------------------------------------------------------------
// Attention pieces. qkvd layout (B, 576, HW): q=ch 0..191, k=192..383, v=384..575.
// ---------------------------------------------------------------------------
__global__ __launch_bounds__(256)
void k_l2norms(const _Float16* __restrict__ qkvd, float* __restrict__ rn) {
  const int r = blockIdx.x;                     // b*384 + isK*192 + ch
  const int b = r / 384;
  const int rem = r - b * 384;
  const int isK = rem / 192;
  const int ch = rem - isK * 192;
  const _Float16* p = qkvd + ((size_t)b * C3 + isK * CDIM + ch) * HW;
  float s = 0.f;
  for (int i = threadIdx.x; i < HW; i += 256) {
    float v = (float)p[i];
    s += v * v;
  }
  __shared__ float red[256];
  red[threadIdx.x] = s;
  __syncthreads();
  for (int st = 128; st > 0; st >>= 1) {
    if (threadIdx.x < st) red[threadIdx.x] += red[threadIdx.x + st];
    __syncthreads();
  }
  if (threadIdx.x == 0) rn[r] = 1.f / fmaxf(sqrtf(red[0]), 1e-12f);
}

// Gram G[d][c] = sum_hw q_d * k_c, scaled by 1/(|q||k|) * temperature[h],
// then row softmax. One workgroup per (b, head).
__global__ __launch_bounds__(256)
void k_gram_softmax(const _Float16* __restrict__ qkvd, const float* __restrict__ rn,
                    const float* __restrict__ temp, float* __restrict__ attnP) {
  const int bh = blockIdx.x;
  const int b = bh >> 3, h = bh & 7;
  const _Float16* q = qkvd + ((size_t)b * C3 + h * 24) * HW;
  const _Float16* k = qkvd + ((size_t)b * C3 + CDIM + h * 24) * HW;
  __shared__ float lq[24 * 132];
  __shared__ float lk[24 * 132];
  __shared__ float G[24 * 24];
  const int tid = threadIdx.x;
  const int p0 = tid, p1 = tid + 256, p2 = tid + 512;
  const int d0 = p0 / 24, c0 = p0 - d0 * 24;
  const int d1 = p1 / 24, c1 = p1 - d1 * 24;
  const int d2 = p2 / 24, c2 = p2 - d2 * 24;
  const bool has2 = (p2 < 576);
  float a0 = 0.f, a1 = 0.f, a2 = 0.f;
  for (int ch = 0; ch < HW; ch += 128) {
    __syncthreads();
    for (int idx = tid; idx < 24 * 128; idx += 256) {
      int d = idx >> 7, e = idx & 127;
      lq[d * 132 + e] = (float)q[(size_t)d * HW + ch + e];
      lk[d * 132 + e] = (float)k[(size_t)d * HW + ch + e];
    }
    __syncthreads();
    const float* q0 = &lq[d0 * 132]; const float* k0p = &lk[c0 * 132];
    const float* q1 = &lq[d1 * 132]; const float* k1p = &lk[c1 * 132];
    const float* q2 = &lq[d2 * 132]; const float* k2p = &lk[c2 * 132];
    for (int e = 0; e < 128; ++e) {
      a0 += q0[e] * k0p[e];
      a1 += q1[e] * k1p[e];
      if (has2) a2 += q2[e] * k2p[e];
    }
  }
  G[p0] = a0;
  G[p1] = a1;
  if (has2) G[p2] = a2;
  __syncthreads();
  if (tid < 24) {
    int d = tid;
    float rq = rn[b * 384 + h * 24 + d];
    float tp = temp[h];
    float row[24];
    float mx = -1e30f;
    for (int c = 0; c < 24; ++c) {
      float v = G[d * 24 + c] * rq * rn[b * 384 + 192 + h * 24 + c] * tp;
      row[c] = v;
      mx = fmaxf(mx, v);
    }
    float s = 0.f;
    for (int c = 0; c < 24; ++c) { row[c] = expf(row[c] - mx); s += row[c]; }
    float inv = 1.f / s;
    for (int c = 0; c < 24; ++c)
      attnP[((size_t)bh * 24 + d) * 24 + c] = row[c] * inv;
  }
}

// out[d][hw] = sum_c attn[d][c] * v[c][hw]  (channel-major f16 result)
__global__ __launch_bounds__(256)
void k_av(const _Float16* __restrict__ qkvd, const float* __restrict__ attnP,
          _Float16* __restrict__ av) {
  const int bh = blockIdx.x;
  const int b = bh >> 3, h = bh & 7;
  __shared__ float la[576];
  for (int i = threadIdx.x; i < 576; i += 256) la[i] = attnP[(size_t)bh * 576 + i];
  __syncthreads();
  const int hw = blockIdx.y * 256 + threadIdx.x;
  const _Float16* v = qkvd + ((size_t)b * C3 + 2 * CDIM + h * 24) * HW + hw;
  float vv[24];
#pragma unroll
  for (int c = 0; c < 24; ++c) vv[c] = (float)v[(size_t)c * HW];
  _Float16* o = av + ((size_t)b * CDIM + h * 24) * HW + hw;
  for (int d = 0; d < 24; ++d) {
    float s = 0.f;
#pragma unroll
    for (int c = 0; c < 24; ++c) s += la[d * 24 + c] * vv[c];
    o[(size_t)d * HW] = (_Float16)s;
  }
}

// ---------------------------------------------------------------------------
// Channel-major (B,Cin,HW) f16 -> pixel-major (NPIX, Cout) f16 transpose with
// zero padding for c in [Cin, Cout). 32-channel x 64-pixel LDS tiles.
// ---------------------------------------------------------------------------
__global__ __launch_bounds__(256)
void k_transpose(const _Float16* __restrict__ in, _Float16* __restrict__ out,
                 int Cin, int Cout) {
  __shared__ _Float16 t[32 * 72];
  const int c0 = blockIdx.x * 32;
  const int p0 = blockIdx.y * 64;
  const int b = p0 >> 14, hw0 = p0 & (HW - 1);
  for (int idx = threadIdx.x; idx < 32 * 64; idx += 256) {
    int cl = idx >> 6, pl = idx & 63;
    int c = c0 + cl;
    _Float16 v = (_Float16)0.f;
    if (c < Cin) v = in[((size_t)b * Cin + c) * HW + hw0 + pl];
    t[cl * 72 + pl] = v;
  }
  __syncthreads();
  for (int idx = threadIdx.x; idx < 64 * 32; idx += 256) {
    int pl = idx >> 5, cl = idx & 31;
    out[(size_t)(p0 + pl) * Cout + c0 + cl] = t[cl * 72 + pl];
  }
}

// ---------------------------------------------------------------------------
extern "C" void kernel_launch(void* const* d_in, const int* in_sizes, int n_in,
                              void* d_out, int out_size, void* d_ws, size_t ws_size,
                              hipStream_t stream) {
  (void)in_sizes; (void)n_in; (void)out_size;
  const float* x      = (const float*)d_in[0];
  const float* temp   = (const float*)d_in[1];
  const float* ln1w   = (const float*)d_in[2];
  const float* ln1b   = (const float*)d_in[3];
  const float* ln2w   = (const float*)d_in[4];
  const float* ln2b   = (const float*)d_in[5];
  const float* w_qkv  = (const float*)d_in[6];
  const float* w_qkvd = (const float*)d_in[7];
  const float* w_proj = (const float*)d_in[8];
  const float* w_in   = (const float*)d_in[9];
  const float* w_dw   = (const float*)d_in[10];
  const float* w_out  = (const float*)d_in[11];
  float* outp = (float*)d_out;

  // Alias-planned workspace (~377 MB peak).
  char* ws = (char*)d_ws;
  const size_t oBig0 = 0;                                   // qkv (f16) then hh (padded 1024 ch)
  const size_t oBig1 = oBig0 + (size_t)NPIX * H2P * 2;      // qkvd (f16) then x1 (f32)
  const size_t oYt   = oBig1 + (size_t)NPIX * C3 * 2;       // y1t / av / y2t
  const size_t oAvt  = oYt   + (size_t)NPIX * CDIM * 2;     // avt
  const size_t oGg   = oAvt  + (size_t)NPIX * CDIM * 2;     // gg
  const size_t oGt   = oGg   + (size_t)NPIX * HID * 2;      // g_t (padded 512)
  const size_t oRn   = oGt   + (size_t)NPIX * HPAD * 2;     // 1536 f32
  const size_t oAt   = oRn   + 4 * 2 * CDIM * 4;            // attn probs
  const size_t oWq   = oAt   + 4 * 8 * 24 * 24 * 4;         // padded f16 weights:
  const size_t oWp   = oWq   + (size_t)C3 * CDIM * 2;       //   qkv 576x192
  const size_t oWi   = oWp   + (size_t)CDIM * CDIM * 2;     //   proj 192x192
  const size_t oWo   = oWi   + (size_t)H2P * CDIM * 2;      //   ffn-in 1024x192
  const size_t need  = oWo   + (size_t)CDIM * HPAD * 2;     //   ffn-out 192x512
  if (ws_size < need) return;

  _Float16* qkv  = (_Float16*)(ws + oBig0);
  _Float16* hh   = (_Float16*)(ws + oBig0);
  _Float16* qkvd = (_Float16*)(ws + oBig1);
  float*    x1   = (float*)   (ws + oBig1);
  _Float16* y1t  = (_Float16*)(ws + oYt);
  _Float16* av   = (_Float16*)(ws + oYt);
  _Float16* y2t  = (_Float16*)(ws + oYt);
  _Float16* avt  = (_Float16*)(ws + oAvt);
  _Float16* gg   = (_Float16*)(ws + oGg);
  _Float16* gt   = (_Float16*)(ws + oGt);
  float*    rn   = (float*)(ws + oRn);
  float*    atp  = (float*)(ws + oAt);
  _Float16* wqh  = (_Float16*)(ws + oWq);
  _Float16* wph  = (_Float16*)(ws + oWp);
  _Float16* wih  = (_Float16*)(ws + oWi);
  _Float16* woh  = (_Float16*)(ws + oWo);

  const dim3 B256(256);
  // ---- Weight prep (f16, zero-padded) ----
  k_wcvt<<<dim3((C3 * CDIM) / 256), B256, 0, stream>>>(w_qkv, wqh, C3, CDIM, CDIM, C3 * CDIM);
  k_wcvt<<<dim3((CDIM * CDIM) / 256), B256, 0, stream>>>(w_proj, wph, CDIM, CDIM, CDIM, CDIM * CDIM);
  k_wcvt<<<dim3((H2P * CDIM) / 256), B256, 0, stream>>>(w_in, wih, HID2, CDIM, CDIM, H2P * CDIM);
  k_wcvt<<<dim3((CDIM * HPAD) / 256), B256, 0, stream>>>(w_out, woh, CDIM, HID, HPAD, CDIM * HPAD);
  // ---- Attention branch ----
  k_ln_to_t<<<dim3(NPIX / 64), B256, 0, stream>>>(x, ln1w, ln1b, y1t);
  k_gemm_wmma<false, CDIM, C3><<<dim3(NPIX / 256, C3 / 64), B256, 0, stream>>>(
      wqh, y1t, qkv, nullptr, nullptr);
  k_dwconv<<<dim3(4 * C3, HW / 256), B256, 0, stream>>>(qkv, w_qkvd, qkvd, C3);
  k_l2norms<<<dim3(4 * 2 * CDIM), B256, 0, stream>>>(qkvd, rn);
  k_gram_softmax<<<dim3(32), B256, 0, stream>>>(qkvd, rn, temp, atp);
  k_av<<<dim3(32, HW / 256), B256, 0, stream>>>(qkvd, atp, av);
  k_transpose<<<dim3(CDIM / 32, NPIX / 64), B256, 0, stream>>>(av, avt, CDIM, CDIM);
  k_gemm_wmma<true, CDIM, CDIM><<<dim3(NPIX / 256, CDIM / 64), B256, 0, stream>>>(
      wph, avt, nullptr, x, x1);
  // ---- FFN branch ----
  k_ln_to_t<<<dim3(NPIX / 64), B256, 0, stream>>>(x1, ln2w, ln2b, y2t);
  k_gemm_wmma<false, CDIM, H2P><<<dim3(NPIX / 256, H2P / 64), B256, 0, stream>>>(
      wih, y2t, hh, nullptr, nullptr);
  k_dwconv_gate<<<dim3(4 * HID, HW / 256), B256, 0, stream>>>(hh, w_dw, gg);
  k_transpose<<<dim3(HPAD / 32, NPIX / 64), B256, 0, stream>>>(gg, gt, HID, HPAD);
  k_gemm_wmma<true, HPAD, CDIM><<<dim3(NPIX / 256, CDIM / 64), B256, 0, stream>>>(
      woh, gt, nullptr, x1, outp);
}
